// CollaborativeLoss_712964571561
// MI455X (gfx1250) — 8-bit matrix operand order repaired and checked by reading, compiled, not run on hardware
//
#include <hip/hip_runtime.h>
#include <stdint.h>

typedef __attribute__((ext_vector_type(8))) int v8i;

#define NSAMP    8192
#define CODE_LEN 128
#define TILE     16
#define JCHUNKS  8

#define NT_CHUNK   ((NSAMP / JCHUNKS) / TILE)
#define LOOP_ITERS ((NT_CHUNK - 4) / 4)
static_assert(NT_CHUNK == 4 + 4 * LOOP_ITERS, "chunk tiling mismatch");

#define TILE_BYTES (TILE * CODE_LEN)

__global__ __launch_bounds__(256)
void prep_kernel(const float* __restrict__ output,
                 const float* __restrict__ codewords,
                 const int*   __restrict__ target,
                 uint8_t*     __restrict__ predB,
                 uint8_t*     __restrict__ cwB,
                 int*         __restrict__ sp,
                 int*         __restrict__ sc,
                 float*       __restrict__ bceRow,
                 int*         __restrict__ sigma)
{
    const int lane = threadIdx.x & 31;
    const int wave = threadIdx.x >> 5;
    const int row  = blockIdx.x * 8 + wave;
    if (row >= NSAMP) return;

    const float4 o4 = ((const float4*)(output + (size_t)row * CODE_LEN))[lane];
    const int    t  = target[row];
    const float4 c4 = ((const float4*)(codewords + (size_t)t * CODE_LEN))[lane];

    const int p0 = o4.x > 0.5f, p1 = o4.y > 0.5f, p2 = o4.z > 0.5f, p3 = o4.w > 0.5f;
    const int c0 = c4.x > 0.5f, c1 = c4.y > 0.5f, c2 = c4.z > 0.5f, c3 = c4.w > 0.5f;

    const uint32_t pw = (uint32_t)p0 | ((uint32_t)p1 << 8) | ((uint32_t)p2 << 16) | ((uint32_t)p3 << 24);
    const uint32_t cw = (uint32_t)c0 | ((uint32_t)c1 << 8) | ((uint32_t)c2 << 16) | ((uint32_t)c3 << 24);
    ((uint32_t*)predB)[(size_t)row * (CODE_LEN / 4) + lane] = pw;
    ((uint32_t*)cwB)[(size_t)row * (CODE_LEN / 4) + lane]   = cw;

    float b = 0.0f;
    b -= c0 ? __logf(o4.x) : log1pf(-o4.x);
    b -= c1 ? __logf(o4.y) : log1pf(-o4.y);
    b -= c2 ? __logf(o4.z) : log1pf(-o4.z);
    b -= c3 ? __logf(o4.w) : log1pf(-o4.w);

    int spp = p0 + p1 + p2 + p3;
    int scp = c0 + c1 + c2 + c3;

    #pragma unroll
    for (int off = 16; off > 0; off >>= 1) {
        b   += __shfl_xor(b,   off, 32);
        spp += __shfl_xor(spp, off, 32);
        scp += __shfl_xor(scp, off, 32);
    }
    if (lane == 0) {
        bceRow[row] = b;
        sp[row]     = spp;
        sc[row]     = scp;
        sigma[row]  = 0x7FFFFFFF;
    }
}

__device__ __forceinline__ v8i tile_acc(const uint8_t* __restrict__ bPtr,
                                        int byteOff, v8i aLo, v8i aHi)
{
    const int4 b0 = *(const int4*)(bPtr + byteOff +  0);
    const int4 b1 = *(const int4*)(bPtr + byteOff + 32);
    const int4 b2 = *(const int4*)(bPtr + byteOff + 64);
    const int4 b3 = *(const int4*)(bPtr + byteOff + 96);
    v8i bLo, bHi;
    bLo[0] = b0.x; bLo[1] = b0.y; bLo[2] = b0.z; bLo[3] = b0.w;
    bLo[4] = b1.x; bLo[5] = b1.y; bLo[6] = b1.z; bLo[7] = b1.w;
    bHi[0] = b2.x; bHi[1] = b2.y; bHi[2] = b2.z; bHi[3] = b2.w;
    bHi[4] = b3.x; bHi[5] = b3.y; bHi[6] = b3.z; bHi[7] = b3.w;
    v8i acc = {0, 0, 0, 0, 0, 0, 0, 0};
    acc = __builtin_amdgcn_wmma_i32_16x16x64_iu8(false, aLo, false, bLo, acc, false, false);
    acc = __builtin_amdgcn_wmma_i32_16x16x64_iu8(false, aHi, false, bHi, acc, false, false);
    return acc;
}

__device__ __forceinline__ void consume_tile(int minv[8], const v8i& acc, int scj)
{
    #pragma unroll
    for (int v = 0; v < 8; ++v) {
        const int val = scj - 2 * acc[v];
        minv[v] = val < minv[v] ? val : minv[v];
    }
}

__global__ __launch_bounds__(32)
void hamming_min_kernel(const uint8_t* __restrict__ predB,
                        const uint8_t* __restrict__ cwB,
                        const int*     __restrict__ sp,
                        const int*     __restrict__ sc,
                        int*           __restrict__ sigma)
{
    const int lane  = threadIdx.x;
    const int iBase = blockIdx.x * TILE;
    const int hi    = lane >> 4;
    const int l16   = lane & 15;
    const int jBegin = blockIdx.y * (NSAMP / JCHUNKS);

    const uint8_t* aRow = predB + (size_t)(iBase + l16) * CODE_LEN + (hi ? 16 : 0);
    const int* ap  = (const int*)aRow;
    const int* ap2 = (const int*)(aRow + 64);
    v8i aLo, aHi;
    aLo[0] = ap[0];   aLo[1] = ap[1];
    aLo[2] = ap[2];   aLo[3] = ap[3];
    aLo[4] = ap[8];   aLo[5] = ap[9];
    aLo[6] = ap[10];  aLo[7] = ap[11];
    aHi[0] = ap2[0];  aHi[1] = ap2[1];
    aHi[2] = ap2[2];  aHi[3] = ap2[3];
    aHi[4] = ap2[8];  aHi[5] = ap2[9];
    aHi[6] = ap2[10]; aHi[7] = ap2[11];

    int minv[8];
    #pragma unroll
    for (int v = 0; v < 8; ++v) minv[v] = 0x7FFFFFFF;

    const uint8_t* bPtr  = cwB + (size_t)(jBegin + l16) * CODE_LEN + (hi ? 16 : 0);
    const int*     scPtr = sc + jBegin + l16;

    v8i accA = tile_acc(bPtr, 0 * TILE_BYTES, aLo, aHi);
    int scA  = scPtr[0 * TILE];
    v8i accB = tile_acc(bPtr, 1 * TILE_BYTES, aLo, aHi);
    int scB  = scPtr[1 * TILE];
    bPtr  += 2 * TILE_BYTES;
    scPtr += 2 * TILE;

    #pragma unroll 1
    for (int it = 0; it < LOOP_ITERS; ++it) {
        v8i accC = tile_acc(bPtr, 0 * TILE_BYTES, aLo, aHi);
        const int scC = scPtr[0 * TILE];
        consume_tile(minv, accA, scA);

        v8i accD = tile_acc(bPtr, 1 * TILE_BYTES, aLo, aHi);
        const int scD = scPtr[1 * TILE];
        consume_tile(minv, accB, scB);

        accA = tile_acc(bPtr, 2 * TILE_BYTES, aLo, aHi);
        scA  = scPtr[2 * TILE];
        consume_tile(minv, accC, scC);

        accB = tile_acc(bPtr, 3 * TILE_BYTES, aLo, aHi);
        scB  = scPtr[3 * TILE];
        consume_tile(minv, accD, scD);

        bPtr  += 4 * TILE_BYTES;
        scPtr += 4 * TILE;
    }

    {
        v8i accC = tile_acc(bPtr, 0 * TILE_BYTES, aLo, aHi);
        const int scC = scPtr[0 * TILE];
        consume_tile(minv, accA, scA);

        v8i accD = tile_acc(bPtr, 1 * TILE_BYTES, aLo, aHi);
        const int scD = scPtr[1 * TILE];
        consume_tile(minv, accB, scB);

        consume_tile(minv, accC, scC);
        consume_tile(minv, accD, scD);
    }
    #pragma unroll
    for (int v = 0; v < 8; ++v) minv[v] += sp[iBase + v + 8 * hi];

    __shared__ int red[32 * 8];
    #pragma unroll
    for (int v = 0; v < 8; ++v) red[lane * 8 + v] = minv[v];
    __syncthreads();
    if (lane < TILE) {
        const int v    = lane & 7;
        const int base = (lane < 8) ? 0 : 16;
        int mn = 0x7FFFFFFF;
        #pragma unroll
        for (int L = 0; L < 16; ++L) {
            const int x = red[(base + L) * 8 + v];
            mn = x < mn ? x : mn;
        }
        atomicMin(&sigma[iBase + lane], mn);
    }
}

__global__ __launch_bounds__(256)
void finalize_kernel(const float* __restrict__ bceRow,
                     const int*   __restrict__ sigma,
                     float*       __restrict__ out)
{
    __shared__ float     fs[256];
    __shared__ long long is[256];
    const int t = threadIdx.x;
    float fb = 0.0f;
    long long si = 0;
    for (int i = t; i < NSAMP; i += 256) {
        fb += bceRow[i];
        si += (long long)sigma[i];
    }
    fs[t] = fb; is[t] = si;
    __syncthreads();
    for (int s = 128; s > 0; s >>= 1) {
        if (t < s) { fs[t] += fs[t + s]; is[t] += is[t + s]; }
        __syncthreads();
    }
    if (t == 0) {
        const double total = (double)fs[0] + (double)CODE_LEN * (double)is[0];
        out[0] = (float)(total / (double)((long long)NSAMP * CODE_LEN));
    }
}

extern "C" void kernel_launch(void* const* d_in, const int* in_sizes, int n_in,
                              void* d_out, int out_size, void* d_ws, size_t ws_size,
                              hipStream_t stream)
{
    const float* output    = (const float*)d_in[0];
    const float* codewords = (const float*)d_in[1];
    const int*   target    = (const int*)d_in[2];
    float*       out       = (float*)d_out;

    uint8_t* ws     = (uint8_t*)d_ws;
    uint8_t* predB  = ws;
    uint8_t* cwB    = ws + (size_t)NSAMP * CODE_LEN;
    int*     sp     = (int*)(ws + 2 * (size_t)NSAMP * CODE_LEN);
    int*     sc     = sp + NSAMP;
    float*   bceRow = (float*)(sc + NSAMP);
    int*     sigma  = (int*)(bceRow + NSAMP);

    prep_kernel<<<NSAMP / 8, 256, 0, stream>>>(output, codewords, target,
                                               predB, cwB, sp, sc, bceRow, sigma);

    dim3 grid(NSAMP / TILE, JCHUNKS);
    hamming_min_kernel<<<grid, 32, 0, stream>>>(predB, cwB, sp, sc, sigma);

    finalize_kernel<<<1, 256, 0, stream>>>(bceRow, sigma, out);
}
